// MolmoAttention_25580825215435
// MI455X (gfx1250) — matrix operand order repaired and checked by reading, not compiled, not run on hardware
//
#include <hip/hip_runtime.h>

// ---------------------------------------------------------------------------
// MolmoAttention for MI455X (gfx1250, wave32, WMMA).
// Pipeline: convert->bf16, QKV GEMM (WMMA bf16), RMSNorm+RoPE, flash causal
// attention (WMMA bf16, online softmax in f32), output-proj GEMM (WMMA bf16).
// Data movement: global_load_async_to_lds_b128 (ASYNCcnt) where no transpose
// is needed, VGPR-transpose staging otherwise, global_prefetch_b8 for next
// tiles.
// ---------------------------------------------------------------------------

#define TT   2048
#define HIDN 2048
#define NH   16
#define DH   128
#define NQKV (3 * HIDN)

typedef __attribute__((ext_vector_type(16))) __bf16 bf16x16;
typedef __attribute__((ext_vector_type(8)))  float  f32x8;

#if defined(__AMDGCN__) && __has_builtin(__builtin_amdgcn_global_load_async_to_lds_b128) && __has_builtin(__builtin_amdgcn_s_wait_asynccnt)
#define USE_ASYNC_COPY 1
#else
#define USE_ASYNC_COPY 0
#endif

#define AS1 __attribute__((address_space(1)))
#define AS3 __attribute__((address_space(3)))

#if USE_ASYNC_COPY
// The async-copy builtin takes v4i* (gcc vector spelling) in AS1 (global src)
// and AS3 (LDS dst). Generic->AS3: LDS offset is the low 32 bits of a generic
// LDS pointer (ISA 10.2); generic->AS1 is
// value-identical, so route both casts through uintptr_t.
typedef int v4i_vs __attribute__((__vector_size__(4 * sizeof(int))));
__device__ __forceinline__ AS3 v4i_vs* lds_cast(void* p) {
  return (AS3 v4i_vs*)(uintptr_t)p;
}
__device__ __forceinline__ AS1 v4i_vs* gbl_cast(const void* p) {
  return (AS1 v4i_vs*)(uintptr_t)p;
}
#endif

__device__ __forceinline__ __bf16 f2bf(float f) {
  unsigned u = __builtin_bit_cast(unsigned, f);
  u += 0x7fffu + ((u >> 16) & 1u);            // round-to-nearest-even
  unsigned short h = (unsigned short)(u >> 16);
  return __builtin_bit_cast(__bf16, h);
}

// A-fragment (16x32 bf16, M x K): lane m=lane&15 holds two 8-elem chunks at
// K = (lane>>4)*8 and K = (lane>>4)*8 + 16 (ISA 7.12.2, 16-bit A layout).
__device__ __forceinline__ bf16x16 frag_a(const __bf16* base, int ldk) {
  int lane = (int)(threadIdx.x & 31);
  const __bf16* p = base + (lane & 15) * ldk + ((lane >> 4) << 3);
  bf16x16 f;
  ((uint4*)&f)[0] = *(const uint4*)(p);
  ((uint4*)&f)[1] = *(const uint4*)(p + 16);
  return f;
}

// B-fragment from B^T (stored [N][K]): lane n=lane&15 holds two 8-elem chunks
// at K = (lane>>4)*8 and K = (lane>>4)*8 + 16 (same K order as frag_a).
__device__ __forceinline__ bf16x16 frag_b(const __bf16* base, int ldk) {
  int lane = (int)(threadIdx.x & 31);
  const __bf16* p = base + (lane & 15) * ldk + ((lane >> 4) << 3);
  bf16x16 f;
  ((uint4*)&f)[0] = *(const uint4*)(p);
  ((uint4*)&f)[1] = *(const uint4*)(p + 16);
  return f;
}

__device__ __forceinline__ f32x8 wmma_bf16(bf16x16 a, bf16x16 b, f32x8 c) {
  return __builtin_amdgcn_wmma_f32_16x16x32_bf16(false, a, false, b,
                                                 (short)0, c, false, false);
}

// ---------------------------------------------------------------------------
// fp32 -> bf16 conversion
// ---------------------------------------------------------------------------
__global__ __launch_bounds__(256) void convert_bf16(const float* __restrict__ src,
                                                    __bf16* __restrict__ dst, int n) {
  int i = (int)(blockIdx.x * 256 + threadIdx.x);
  if (i < n) dst[i] = f2bf(src[i]);
}

// ---------------------------------------------------------------------------
// C[M][N] (f32) = A[M][K] (bf16) * B[K][N] (bf16)
// block tile 128x128, 8 waves (4x2), wave tile 32x64, K-step 32.
// A tile staged via async global->LDS copy; B tile transposed through VGPRs.
// ---------------------------------------------------------------------------
__global__ __launch_bounds__(256) void gemm_bf16(const __bf16* __restrict__ A,
                                                 const __bf16* __restrict__ B,
                                                 float* __restrict__ C,
                                                 int M, int N, int K) {
  __shared__ __align__(16) __bf16 As[128 * 40];  // [m][k], ld=40 (pad 8)
  __shared__ __align__(16) __bf16 Bt[128 * 40];  // [n][k], ld=40

  const int tid  = (int)threadIdx.x;
  const int wave = tid >> 5;
  const int lane = tid & 31;
  const int wm = wave & 3;   // 4 wave-rows * 32 = 128
  const int wn = wave >> 2;  // 2 wave-cols * 64 = 128
  const int mBase = (int)blockIdx.y * 128;
  const int nBase = (int)blockIdx.x * 128;

  f32x8 acc[2][4] = {};

  for (int kb = 0; kb < K; kb += 32) {
    __syncthreads();
    // Stage A tile 128x32 (2 x 16B per thread): direct global->LDS copy.
#pragma unroll
    for (int t = 0; t < 2; ++t) {
      int idx = tid + t * 256;
      int m = idx >> 2, kc = (idx & 3) << 3;
#if USE_ASYNC_COPY
      __builtin_amdgcn_global_load_async_to_lds_b128(
          gbl_cast(A + (size_t)(mBase + m) * K + kb + kc),
          lds_cast(As + m * 40 + kc), 0, 0);
#else
      uint4 v = *(const uint4*)(A + (size_t)(mBase + m) * K + kb + kc);
      *(uint4*)(As + m * 40 + kc) = v;
#endif
    }
    // Stage B tile 32x128, transposed into Bt[n][k] (needs VGPR bounce).
#pragma unroll
    for (int t = 0; t < 2; ++t) {
      int idx = tid + t * 256;
      int k = idx >> 4, nc = (idx & 15) << 3;
      uint4 v = *(const uint4*)(B + (size_t)(kb + k) * N + nBase + nc);
      const __bf16* e = (const __bf16*)&v;
#pragma unroll
      for (int i = 0; i < 8; ++i) Bt[(nc + i) * 40 + k] = e[i];
    }
    // Prefetch next K tile (global_prefetch_b8)
    if (kb + 32 < K) {
      int m = tid >> 2, kc = (tid & 3) << 3;
      __builtin_prefetch(A + (size_t)(mBase + m) * K + kb + 32 + kc, 0, 3);
      int k2 = tid >> 4, nc = (tid & 15) << 3;
      __builtin_prefetch(B + (size_t)(kb + 32 + k2) * N + nBase + nc, 0, 3);
    }
#if USE_ASYNC_COPY
    __builtin_amdgcn_s_wait_asynccnt(0);
#endif
    __syncthreads();

    bf16x16 af[2], bfr[4];
    af[0] = frag_a(As + (wm * 32 + 0)  * 40, 40);
    af[1] = frag_a(As + (wm * 32 + 16) * 40, 40);
#pragma unroll
    for (int j = 0; j < 4; ++j)
      bfr[j] = frag_b(Bt + (wn * 64 + j * 16) * 40, 40);
#pragma unroll
    for (int i = 0; i < 2; ++i)
#pragma unroll
      for (int j = 0; j < 4; ++j)
        acc[i][j] = wmma_bf16(af[i], bfr[j], acc[i][j]);
  }

  // Epilogue: C layout — VGPR e, lane: M = e + 8*(lane>>4), N = lane&15
  const int col0 = lane & 15, rh = lane >> 4;
#pragma unroll
  for (int i = 0; i < 2; ++i)
#pragma unroll
    for (int j = 0; j < 4; ++j)
#pragma unroll
      for (int e = 0; e < 8; ++e) {
        int row = mBase + wm * 32 + i * 16 + e + 8 * rh;
        int col = nBase + wn * 64 + j * 16 + col0;
        C[(size_t)row * N + col] = acc[i][j][e];
      }
}

// ---------------------------------------------------------------------------
// Per-token RMSNorm (over 2048) on Q/K, RoPE per head (D=128, theta=1e6),
// pack Q/K/V to bf16 in [T][H*D] layout.
// ---------------------------------------------------------------------------
__global__ __launch_bounds__(256) void norm_rope(const float* __restrict__ qkv,
                                                 const int* __restrict__ pos,
                                                 const float* __restrict__ qw,
                                                 const float* __restrict__ kw,
                                                 __bf16* __restrict__ Qb,
                                                 __bf16* __restrict__ Kb,
                                                 __bf16* __restrict__ Vb) {
  const int t = (int)blockIdx.x, tid = (int)threadIdx.x;
  const float* q = qkv + (size_t)t * NQKV;
  const float* k = q + HIDN;
  const float* v = q + 2 * HIDN;
  __shared__ float red[256];

  float sq = 0.f, sk = 0.f;
  for (int i = tid; i < HIDN; i += 256) {
    float a = q[i]; sq += a * a;
    float b = k[i]; sk += b * b;
  }
  red[tid] = sq; __syncthreads();
  for (int s2 = 128; s2 > 0; s2 >>= 1) { if (tid < s2) red[tid] += red[tid + s2]; __syncthreads(); }
  const float rq = rsqrtf(red[0] * (1.0f / HIDN) + 1e-5f);
  __syncthreads();
  red[tid] = sk; __syncthreads();
  for (int s2 = 128; s2 > 0; s2 >>= 1) { if (tid < s2) red[tid] += red[tid + s2]; __syncthreads(); }
  const float rk = rsqrtf(red[0] * (1.0f / HIDN) + 1e-5f);

  const float p = (float)pos[t];
  const float LOG_THETA_OVER_HALF = 0.21586735f;  // ln(1e6)/64
  for (int pr = tid; pr < NH * (DH / 2); pr += 256) {
    int h = pr >> 6, d = pr & 63;
    int i1 = h * DH + d, i2 = i1 + 64;
    float ang = p * __expf(-(float)d * LOG_THETA_OVER_HALF);
    float c = cosf(ang), s = sinf(ang);
    float q1 = q[i1] * rq * qw[i1], q2 = q[i2] * rq * qw[i2];
    Qb[(size_t)t * HIDN + i1] = f2bf(q1 * c - q2 * s);
    Qb[(size_t)t * HIDN + i2] = f2bf(q2 * c + q1 * s);
    float k1 = k[i1] * rk * kw[i1], k2 = k[i2] * rk * kw[i2];
    Kb[(size_t)t * HIDN + i1] = f2bf(k1 * c - k2 * s);
    Kb[(size_t)t * HIDN + i2] = f2bf(k2 * c + k1 * s);
  }
  for (int i = tid; i < HIDN; i += 256) Vb[(size_t)t * HIDN + i] = f2bf(v[i]);
}

// ---------------------------------------------------------------------------
// Flash-style causal attention. Block = (head, 128 q-rows); 8 waves x 16 rows.
// KV tiles of 32; S=Q*K^T and O+=P*V via bf16 WMMA; softmax state in f32.
// K tile staged via async global->LDS; V transposed through VGPRs.
// ---------------------------------------------------------------------------
__global__ __launch_bounds__(256) void flash_attn(const __bf16* __restrict__ Q,
                                                  const __bf16* __restrict__ Kg,
                                                  const __bf16* __restrict__ Vg,
                                                  __bf16* __restrict__ Ob) {
  __shared__ __align__(16) __bf16 Ks[32 * 136];     // [kv][d], ld=136
  __shared__ __align__(16) __bf16 Vt[128 * 40];     // [d][kv], ld=40
  __shared__ __align__(16) __bf16 Ps[8 * 16 * 40];  // per-wave P tile [16][32]

  const int tid = (int)threadIdx.x, lane = tid & 31, wave = tid >> 5;
  const int h = (int)blockIdx.y;
  const int qb = (int)blockIdx.x * 128;
  const int col0 = lane & 15, rh = lane >> 4;
  const float scale = 0.08838834764831845f;  // 1/sqrt(128)

  // Q fragments for this wave's 16 rows (kept in registers all kernel).
  bf16x16 qf[4];
  {
    const int qrow = qb + wave * 16 + (lane & 15);
#pragma unroll
    for (int kk = 0; kk < 4; ++kk) {
      const __bf16* p = Q + (size_t)qrow * HIDN + h * DH + kk * 32 + ((lane >> 4) << 3);
      ((uint4*)&qf[kk])[0] = *(const uint4*)(p);
      ((uint4*)&qf[kk])[1] = *(const uint4*)(p + 16);
    }
  }

  f32x8 o[8] = {};
  float mrow[8], lrow[8];
#pragma unroll
  for (int e = 0; e < 8; ++e) { mrow[e] = -INFINITY; lrow[e] = 0.f; }

  const int nkv = qb + 128;  // causal bound for this q block
  for (int kvb = 0; kvb < nkv; kvb += 32) {
    __syncthreads();
    // Stage K rows (row-major, direct copy) and V rows (transposed).
#pragma unroll
    for (int t = 0; t < 2; ++t) {
      int idx = tid + t * 256;
      int r = idx >> 4, dc = (idx & 15) << 3;
#if USE_ASYNC_COPY
      __builtin_amdgcn_global_load_async_to_lds_b128(
          gbl_cast(Kg + (size_t)(kvb + r) * HIDN + h * DH + dc),
          lds_cast(Ks + r * 136 + dc), 0, 0);
#else
      uint4 kv4 = *(const uint4*)(Kg + (size_t)(kvb + r) * HIDN + h * DH + dc);
      *(uint4*)(Ks + r * 136 + dc) = kv4;
#endif
      uint4 vv4 = *(const uint4*)(Vg + (size_t)(kvb + r) * HIDN + h * DH + dc);
      const __bf16* e8 = (const __bf16*)&vv4;
#pragma unroll
      for (int i = 0; i < 8; ++i) Vt[(dc + i) * 40 + r] = e8[i];
    }
    if (kvb + 32 < nkv) {
      int r = tid >> 4, dc = (tid & 15) << 3;
      __builtin_prefetch(Kg + (size_t)(kvb + 32 + r) * HIDN + h * DH + dc, 0, 3);
      __builtin_prefetch(Vg + (size_t)(kvb + 32 + r) * HIDN + h * DH + dc, 0, 3);
    }
#if USE_ASYNC_COPY
    __builtin_amdgcn_s_wait_asynccnt(0);
#endif
    __syncthreads();

    // S = Q * K^T  (two 16-col subtiles)
    f32x8 s[2];
#pragma unroll
    for (int j = 0; j < 2; ++j) {
      f32x8 a = {};
#pragma unroll
      for (int kk = 0; kk < 4; ++kk)
        a = wmma_bf16(qf[kk], frag_b(Ks + (j * 16) * 136 + kk * 32, 136), a);
      s[j] = a;
    }
    // scale + causal mask
#pragma unroll
    for (int j = 0; j < 2; ++j)
#pragma unroll
      for (int e = 0; e < 8; ++e) {
        int kv = kvb + j * 16 + col0;
        int qr = qb + wave * 16 + e + 8 * rh;
        s[j][e] = (kv <= qr) ? s[j][e] * scale : -INFINITY;
      }
    // Online softmax (row stats replicated across each 16-lane group).
#pragma unroll
    for (int e = 0; e < 8; ++e) {
      float mx = fmaxf(s[0][e], s[1][e]);
#pragma unroll
      for (int off = 1; off < 16; off <<= 1) mx = fmaxf(mx, __shfl_xor(mx, off, 32));
      float nm = fmaxf(mrow[e], mx);
      float alpha = __expf(mrow[e] - nm);
      float p0 = __expf(s[0][e] - nm);
      float p1 = __expf(s[1][e] - nm);
      float rs = p0 + p1;
#pragma unroll
      for (int off = 1; off < 16; off <<= 1) rs += __shfl_xor(rs, off, 32);
      lrow[e] = lrow[e] * alpha + rs;
      mrow[e] = nm;
      s[0][e] = p0; s[1][e] = p1;
#pragma unroll
      for (int n = 0; n < 8; ++n) o[n][e] *= alpha;
    }
    // P (C-layout) -> per-wave LDS -> A-layout fragment.
    __bf16* pw = Ps + wave * 16 * 40;
#pragma unroll
    for (int e = 0; e < 8; ++e) {
      int row = e + 8 * rh;
      pw[row * 40 + col0]      = f2bf(s[0][e]);
      pw[row * 40 + 16 + col0] = f2bf(s[1][e]);
    }
    __syncthreads();
    bf16x16 pf = frag_a(pw, 40);
#pragma unroll
    for (int n = 0; n < 8; ++n)
      o[n] = wmma_bf16(pf, frag_b(Vt + (n * 16) * 40, 40), o[n]);
  }

  // Normalize and store O as bf16 in [T][H*D].
#pragma unroll
  for (int n = 0; n < 8; ++n)
#pragma unroll
    for (int e = 0; e < 8; ++e) {
      int qr = qb + wave * 16 + e + 8 * rh;
      int d  = n * 16 + col0;
      Ob[(size_t)qr * HIDN + h * DH + d] = f2bf(o[n][e] / lrow[e]);
    }
}

// ---------------------------------------------------------------------------
extern "C" void kernel_launch(void* const* d_in, const int* in_sizes, int n_in,
                              void* d_out, int out_size, void* d_ws, size_t ws_size,
                              hipStream_t stream) {
  (void)in_sizes; (void)n_in; (void)out_size; (void)ws_size;
  const float* hidden    = (const float*)d_in[0];
  const int*   positions = (const int*)d_in[1];
  const float* w_qkv     = (const float*)d_in[2];
  const float* q_norm_w  = (const float*)d_in[3];
  const float* k_norm_w  = (const float*)d_in[4];
  const float* w_o       = (const float*)d_in[5];
  float* out = (float*)d_out;

  char* ws = (char*)d_ws;
  __bf16* hid_bf  = (__bf16*)(ws + ((size_t)0   << 20));  //  8 MB
  __bf16* wqkv_bf = (__bf16*)(ws + ((size_t)8   << 20));  // 24 MB
  __bf16* wo_bf   = (__bf16*)(ws + ((size_t)32  << 20));  //  8 MB
  float*  qkv     = (float*) (ws + ((size_t)40  << 20));  // 48 MB
  __bf16* attn_bf = (__bf16*)(ws + ((size_t)40  << 20));  // reuses qkv (8 MB)
  __bf16* Qb      = (__bf16*)(ws + ((size_t)88  << 20));  //  8 MB
  __bf16* Kb      = (__bf16*)(ws + ((size_t)96  << 20));  //  8 MB
  __bf16* Vb      = (__bf16*)(ws + ((size_t)104 << 20));  //  8 MB

  const int nHid  = TT * HIDN;        // 4,194,304
  const int nWqkv = HIDN * NQKV;      // 12,582,912

  convert_bf16<<<(nHid  + 255) / 256, 256, 0, stream>>>(hidden, hid_bf, nHid);
  convert_bf16<<<(nWqkv + 255) / 256, 256, 0, stream>>>(w_qkv, wqkv_bf, nWqkv);
  convert_bf16<<<(nHid  + 255) / 256, 256, 0, stream>>>(w_o, wo_bf, nHid);

  // QKV projection: [2048,2048] x [2048,6144] -> f32
  gemm_bf16<<<dim3(NQKV / 128, TT / 128), 256, 0, stream>>>(
      hid_bf, wqkv_bf, qkv, TT, NQKV, HIDN);

  // RMSNorm + RoPE + bf16 packing of Q/K/V
  norm_rope<<<TT, 256, 0, stream>>>(qkv, positions, q_norm_w, k_norm_w, Qb, Kb, Vb);

  // Causal flash attention: grid = (q-tiles, heads)
  flash_attn<<<dim3(TT / 128, NH), 256, 0, stream>>>(Qb, Kb, Vb, attn_bf);

  // Output projection: [2048,2048] x [2048,2048] -> d_out f32
  gemm_bf16<<<dim3(HIDN / 128, TT / 128), 256, 0, stream>>>(
      attn_bf, wo_bf, out, TT, HIDN, HIDN);
}
